// UltraOptimizedINLBlock_3599182594271
// MI455X (gfx1250) — matrix operand order repaired and checked by reading, not compiled, not run on hardware
//
#include <hip/hip_runtime.h>
#include <cstdint>
#include <cstddef>

// ---------------------------------------------------------------------------
// Problem constants (from reference): B=2, S=2048, D=1024, H=16, HD=64, FF=4096
// ---------------------------------------------------------------------------
#define B_  2
#define S_  2048
#define D_  1024
#define H_  16
#define HD_ 64
#define FF_ 4096
#define N_  (B_*S_)
#define DT_ 0.1f

typedef __bf16 bf16_t;
typedef __attribute__((ext_vector_type(16))) __bf16 v16bf;
typedef __attribute__((ext_vector_type(8)))  __bf16 v8bf;
typedef __attribute__((ext_vector_type(8)))  float  v8f;

union Frag16 { v16bf v; v8bf h[2]; };

__device__ __forceinline__ v8f wmma_bf16f32(v16bf a, v16bf b, v8f c) {
  // D(f32 16x16) = A(bf16 16x32) * B(bf16 32x16) + C
  return __builtin_amdgcn_wmma_f32_16x16x32_bf16(false, a, false, b, (short)0, c,
                                                 false, false);
}

// Async DMA: global -> LDS, 16B per lane, tracked by ASYNCcnt (no VGPR round-trip).
__device__ __forceinline__ void async_ld_b128(uint32_t lds_off, const void* g) {
  asm volatile("global_load_async_to_lds_b128 %0, %1, off"
               :: "v"(lds_off), "v"((uint64_t)(uintptr_t)g) : "memory");
}

// ---------------------------------------------------------------------------
// Generic bf16 WMMA GEMM:  out[M,Nn] = A[M,K] @ B[K,Nn] (+bias) (+resid)
// mode 0: fp32 store (optional residual add); 1: bf16 store; 2: GELU -> bf16
// Block = 256 threads (8 waves). Tile 128x128, K-step 32.
// Double-buffered LDS staging via GLOBAL_LOAD_ASYNC_TO_LDS_B128 (ASYNCcnt):
// tile k+1 DMA overlaps the 8-WMMA burst on tile k.
// ---------------------------------------------------------------------------
#define TM 128
#define TN 128
#define TK 32
#define LDA 40     // padded LDS stride for A (conflict-free b128 frag loads)
#define LDB 136    // padded LDS stride for B

__global__ __launch_bounds__(256) void gemm_bf16_wmma(
    const bf16_t* __restrict__ A, const bf16_t* __restrict__ Bm,
    const float* __restrict__ bias, const float* __restrict__ resid,
    float* __restrict__ outF, bf16_t* __restrict__ outB,
    int M, int Nn, int K, int mode)
{
  __shared__ bf16_t As[2][TM * LDA];
  __shared__ bf16_t Bs[2][TK * LDB];

  const int t   = threadIdx.x;
  const int l   = t & 31;
  const int wid = t >> 5;
  const int wr  = wid >> 1;   // 0..3 (row of wave grid)
  const int wc  = wid & 1;    // 0..1 (col of wave grid)
  const int lnn = l & 15;
  const int kh  = l >> 4;
  const int m_blk = blockIdx.y * TM;
  const int n_blk = blockIdx.x * TN;

  v8f acc[2][4] = {};

  const int ar0 = t >> 2;         // 0..63  (A tile row, 2 passes)
  const int akg = (t & 3) * 8;    // 0,8,16,24
  const int br0 = t >> 4;         // 0..15  (B tile row, 2 passes)
  const int bng = (t & 15) * 8;   // 0..120

  // Per-thread LDS destinations (low 32 bits of flat pointer = LDS byte addr)
  const uint32_t lds_a0[2] = {
    (uint32_t)(uintptr_t)(&As[0][0] + ar0 * LDA + akg),
    (uint32_t)(uintptr_t)(&As[1][0] + ar0 * LDA + akg) };
  const uint32_t lds_a1[2] = {
    (uint32_t)(uintptr_t)(&As[0][0] + (ar0 + 64) * LDA + akg),
    (uint32_t)(uintptr_t)(&As[1][0] + (ar0 + 64) * LDA + akg) };
  const uint32_t lds_b0[2] = {
    (uint32_t)(uintptr_t)(&Bs[0][0] + br0 * LDB + bng),
    (uint32_t)(uintptr_t)(&Bs[1][0] + br0 * LDB + bng) };
  const uint32_t lds_b1[2] = {
    (uint32_t)(uintptr_t)(&Bs[0][0] + (br0 + 16) * LDB + bng),
    (uint32_t)(uintptr_t)(&Bs[1][0] + (br0 + 16) * LDB + bng) };

  auto stage = [&](int buf, int k0) {
    async_ld_b128(lds_a0[buf], A + (size_t)(m_blk + ar0)      * K + k0 + akg);
    async_ld_b128(lds_a1[buf], A + (size_t)(m_blk + ar0 + 64) * K + k0 + akg);
    async_ld_b128(lds_b0[buf], Bm + (size_t)(k0 + br0)      * Nn + n_blk + bng);
    async_ld_b128(lds_b1[buf], Bm + (size_t)(k0 + br0 + 16) * Nn + n_blk + bng);
  };

  const int nk = K / TK;
  stage(0, 0);                             // prologue: prefetch tile 0

  for (int kk = 0; kk < nk; ++kk) {
    const int p = kk & 1;
    if (kk + 1 < nk) {
      stage(p ^ 1, (kk + 1) * TK);         // prefetch next tile into other buffer
      asm volatile("s_wait_asynccnt 0x4" ::: "memory");  // tile kk's 4 DMAs done
    } else {
      asm volatile("s_wait_asynccnt 0x0" ::: "memory");
    }
    __syncthreads();                        // all waves' DMA data visible

    Frag16 a[2];
    #pragma unroll
    for (int i = 0; i < 2; ++i) {  // A frag: lane holds row m, K pairs per VGPR
      const int ml = wr * 32 + i * 16 + lnn;
      a[i].h[0] = *(const v8bf*)(&As[p][0] + ml * LDA + kh * 8);
      a[i].h[1] = *(const v8bf*)(&As[p][0] + ml * LDA + 16 + kh * 8);
    }
    #pragma unroll
    for (int j = 0; j < 4; ++j) {  // B frag: lane holds col N=lnn, 16 K values (table order)
      Frag16 b;
      const int nl = wc * 64 + j * 16;
      #pragma unroll
      for (int e = 0; e < 16; ++e) {
        const int kr = (e < 8) ? kh * 8 + e : 16 + kh * 8 + (e - 8);
        b.v[e] = Bs[p][kr * LDB + nl + lnn];
      }
      #pragma unroll
      for (int i = 0; i < 2; ++i)
        acc[i][j] = wmma_bf16f32(a[i].v, b.v, acc[i][j]);
    }
    __syncthreads();   // reads of buffer p done before next DMA overwrites it
  }

  // Epilogue. C frag: VGPR r, lane l -> row r+8*(l>>4), col l&15.
  #pragma unroll
  for (int i = 0; i < 2; ++i)
    #pragma unroll
    for (int j = 0; j < 4; ++j)
      #pragma unroll
      for (int r = 0; r < 8; ++r) {
        const int row = m_blk + wr * 32 + i * 16 + r + 8 * kh;
        const int col = n_blk + wc * 64 + j * 16 + lnn;
        float v = acc[i][j][r];
        if (bias) v += bias[col];
        const size_t idx = (size_t)row * Nn + col;
        if (mode == 2) {
          float g = 0.5f * v * (1.0f + erff(v * 0.70710678118654752f));
          outB[idx] = (bf16_t)g;
        } else if (mode == 1) {
          outB[idx] = (bf16_t)v;
        } else {
          if (resid) v += resid[idx];
          outF[idx] = v;
        }
      }
}

// ---------------------------------------------------------------------------
// Causal multi-head attention, flash-style online softmax.
// One wave (32 threads) per (b, h, 16-query tile). Key tiles of 32.
// Q@K^T via pre-transposed K buffer ktr[B,H,HD,S] (contiguous B-frags).
// P converted C-layout -> A-layout through LDS scratch.
// ---------------------------------------------------------------------------
__global__ __launch_bounds__(32) void attn_causal_wmma(
    const bf16_t* __restrict__ qkv,   // [N_, 3*D_]  (q | k | v)
    const bf16_t* __restrict__ ktr,   // [B_,H_,HD_,S_]
    bf16_t* __restrict__ outh)        // [N_, D_]
{
  __shared__ bf16_t Pls[16 * 40];
  const int l   = threadIdx.x;
  const int lnn = l & 15;
  const int kh  = l >> 4;
  const int q0  = blockIdx.x * 16;
  const int b   = blockIdx.y / H_;
  const int h   = blockIdx.y % H_;

  Frag16 qf[2];                        // Q[16 x 64] as two 16x32 A-frags
  #pragma unroll
  for (int f = 0; f < 2; ++f) {
    const bf16_t* qrow = qkv + (size_t)(b * S_ + q0 + lnn) * (3 * D_) + h * HD_ + f * 32;
    qf[f].h[0] = *(const v8bf*)(qrow + kh * 8);
    qf[f].h[1] = *(const v8bf*)(qrow + 16 + kh * 8);
  }

  v8f acc[4] = {};
  float rmax[8], rsum[8];
  #pragma unroll
  for (int r = 0; r < 8; ++r) { rmax[r] = -3.0e38f; rsum[r] = 0.0f; }

  const int nkt = (q0 + 47) / 32;      // key tiles needed (causal)
  const bf16_t* ktbase = ktr + (size_t)blockIdx.y * HD_ * S_;

  for (int kt = 0; kt < nkt; ++kt) {
    const int k0 = kt * 32;
    v8f s[2] = {};                     // scores 16q x 32k as two 16x16 frags
    #pragma unroll
    for (int c = 0; c < 2; ++c)
      #pragma unroll
      for (int f = 0; f < 2; ++f) {
        Frag16 kb;                     // B frag: lane N = key, 16 K (hd) values
        #pragma unroll
        for (int e = 0; e < 16; ++e) {
          const int kr = (e < 8) ? kh * 8 + e : 16 + kh * 8 + (e - 8);
          kb.v[e] = ktbase[(size_t)(f * 32 + kr) * S_ + k0 + c * 16 + lnn];
        }
        s[c] = wmma_bf16f32(qf[f].v, kb.v, s[c]);
      }
    #pragma unroll
    for (int c = 0; c < 2; ++c)        // scale + causal mask
      #pragma unroll
      for (int r = 0; r < 8; ++r) {
        float sv = s[c][r] * 0.125f;   // 1/sqrt(64)
        const int q  = q0 + r + 8 * kh;
        const int ki = k0 + c * 16 + lnn;
        s[c][r] = (ki > q) ? -1.0e30f : sv;
      }
    float nm[8], fac[8];               // online softmax (rows split by lane half)
    #pragma unroll
    for (int r = 0; r < 8; ++r) {
      float m = fmaxf(s[0][r], s[1][r]);
      #pragma unroll
      for (int off = 1; off < 16; off <<= 1)
        m = fmaxf(m, __shfl_xor(m, off, 16));
      nm[r]   = fmaxf(rmax[r], m);
      fac[r]  = expf(rmax[r] - nm[r]);
      rmax[r] = nm[r];
    }
    #pragma unroll
    for (int j = 0; j < 4; ++j)
      #pragma unroll
      for (int r = 0; r < 8; ++r)
        acc[j][r] *= fac[r];
    #pragma unroll
    for (int c = 0; c < 2; ++c)
      #pragma unroll
      for (int r = 0; r < 8; ++r) {
        float p = expf(s[c][r] - nm[r]);
        s[c][r] = p;
        Pls[(r + 8 * kh) * 40 + c * 16 + lnn] = (bf16_t)p;
      }
    #pragma unroll
    for (int r = 0; r < 8; ++r) {
      float sm = s[0][r] + s[1][r];
      #pragma unroll
      for (int off = 1; off < 16; off <<= 1)
        sm += __shfl_xor(sm, off, 16);
      rsum[r] = rsum[r] * fac[r] + sm;
    }
    asm volatile("s_wait_dscnt 0x0" ::: "memory");  // P visible for A-frag reload
    Frag16 pa;                          // P[16x32] as A-frag
    pa.h[0] = *(const v8bf*)(Pls + lnn * 40 + kh * 8);
    pa.h[1] = *(const v8bf*)(Pls + lnn * 40 + 16 + kh * 8);
    #pragma unroll
    for (int j = 0; j < 4; ++j) {       // acc += P @ V[32 x 64]
      Frag16 vb;                        // lane N = head dim lnn, 16 K (key) values
      #pragma unroll
      for (int e = 0; e < 16; ++e) {
        const int kr = (e < 8) ? kh * 8 + e : 16 + kh * 8 + (e - 8);
        vb.v[e] = qkv[(size_t)(b * S_ + k0 + kr) * (3 * D_) + 2 * D_ + h * HD_ + j * 16 + lnn];
      }
      acc[j] = wmma_bf16f32(pa.v, vb.v, acc[j]);
    }
  }
  #pragma unroll
  for (int j = 0; j < 4; ++j)
    #pragma unroll
    for (int r = 0; r < 8; ++r) {
      const int row = q0 + r + 8 * kh;
      outh[(size_t)(b * S_ + row) * D_ + h * HD_ + j * 16 + lnn] =
          (bf16_t)(acc[j][r] / rsum[r]);
    }
}

// ---------------------------------------------------------------------------
// Elementwise / LN / transpose kernels
// ---------------------------------------------------------------------------
__global__ __launch_bounds__(256) void layernorm_f32(
    const float* __restrict__ x, const float* __restrict__ g,
    const float* __restrict__ bt, float* __restrict__ yf, bf16_t* __restrict__ yb)
{
  __shared__ float ssum[256], ssq[256];
  const int t = threadIdx.x;
  const size_t base = (size_t)blockIdx.x * D_;
  float s = 0.f, q = 0.f;
  for (int i = t; i < D_; i += 256) { float v = x[base + i]; s += v; q += v * v; }
  ssum[t] = s; ssq[t] = q;
  __syncthreads();
  for (int st = 128; st > 0; st >>= 1) {
    if (t < st) { ssum[t] += ssum[t + st]; ssq[t] += ssq[t + st]; }
    __syncthreads();
  }
  const float mean = ssum[0] * (1.0f / D_);
  const float var  = ssq[0] * (1.0f / D_) - mean * mean;
  const float inv  = rsqrtf(var + 1e-5f);
  for (int i = t; i < D_; i += 256) {
    float v = (x[base + i] - mean) * inv * g[i] + bt[i];
    if (yf) yf[base + i] = v;
    if (yb) yb[base + i] = (bf16_t)v;
  }
}

__global__ __launch_bounds__(256) void cast_f32_bf16(
    const float* __restrict__ in, bf16_t* __restrict__ out, int n)
{
  int i = blockIdx.x * 256 + threadIdx.x;
  if (i < n) out[i] = (bf16_t)in[i];
}

__global__ __launch_bounds__(256) void transpose_k_kern(
    const bf16_t* __restrict__ qkv, bf16_t* __restrict__ ktr)
{
  const size_t idx = (size_t)blockIdx.x * 256 + threadIdx.x; // over B*H*HD*S
  const int s  = (int)(idx % S_);
  size_t r = idx / S_;
  const int hd = (int)(r % HD_); r /= HD_;
  const int h  = (int)(r % H_);
  const int b  = (int)(r / H_);
  ktr[idx] = qkv[(size_t)(b * S_ + s) * (3 * D_) + D_ + h * HD_ + hd];
}

__global__ __launch_bounds__(256) void resid_cast_kern(
    const float* __restrict__ x, const float* __restrict__ ap,
    float* __restrict__ x1, bf16_t* __restrict__ ctxb, int n)
{
  int i = blockIdx.x * 256 + threadIdx.x;
  if (i < n) { float a = ap[i]; x1[i] = x[i] + a; ctxb[i] = (bf16_t)a; }
}

__global__ __launch_bounds__(256) void inl_update_kern(
    const float* __restrict__ cpre, const float* __restrict__ wxo,
    float* __restrict__ xs, float* __restrict__ vs, bf16_t* __restrict__ xsb, int n)
{
  int i = blockIdx.x * 256 + threadIdx.x;
  if (i < n) {
    float u  = tanhf(cpre[i] + wxo[i]);
    float vn = vs[i] + DT_ * (u - xs[i] - vs[i]);
    float xn = xs[i] + DT_ * vn;
    vs[i] = vn; xs[i] = xn; xsb[i] = (bf16_t)xn;
  }
}

__global__ __launch_bounds__(256) void add_inplace_kern(
    float* __restrict__ a, const float* __restrict__ b, int n)
{
  int i = blockIdx.x * 256 + threadIdx.x;
  if (i < n) a[i] += b[i];
}

// ---------------------------------------------------------------------------
// Host orchestration
// ---------------------------------------------------------------------------
extern "C" void kernel_launch(void* const* d_in, const int* in_sizes, int n_in,
                              void* d_out, int out_size, void* d_ws, size_t ws_size,
                              hipStream_t stream) {
  (void)in_sizes; (void)n_in; (void)out_size; (void)ws_size;
  const float* x      = (const float*)d_in[0];
  const float* qkv_w  = (const float*)d_in[1];
  const float* qkv_b  = (const float*)d_in[2];
  const float* out_w  = (const float*)d_in[3];
  const float* out_b  = (const float*)d_in[4];
  const float* g_attn = (const float*)d_in[5];
  const float* b_attn = (const float*)d_in[6];
  const float* g1     = (const float*)d_in[7];
  const float* b1     = (const float*)d_in[8];
  const float* g2     = (const float*)d_in[9];
  const float* b2     = (const float*)d_in[10];
  const float* w_ctx  = (const float*)d_in[11];
  const float* w_x    = (const float*)d_in[12];
  const float* b_inl  = (const float*)d_in[13];
  const float* ff_w1  = (const float*)d_in[14];
  const float* ff_b1  = (const float*)d_in[15];
  const float* ff_w2  = (const float*)d_in[16];
  const float* ff_b2  = (const float*)d_in[17];
  float* out = (float*)d_out;

  char* wsp = (char*)d_ws;
  size_t cur = 0;
  auto alloc = [&](size_t bytes) -> void* {
    void* p = wsp + cur;
    cur += (bytes + 255) & ~(size_t)255;
    return p;
  };

  const size_t ND  = (size_t)N_ * D_;        // 4,194,304
  const size_t NFF = (size_t)N_ * FF_;       // 16,777,216
  const size_t N3D = (size_t)N_ * 3 * D_;    // 12,582,912
  const size_t bufA_elems = (N3D > NFF) ? N3D : NFF;

  // bf16 weight copies
  bf16_t* qkvw_b = (bf16_t*)alloc((size_t)D_ * 3 * D_ * sizeof(bf16_t));
  bf16_t* outw_b = (bf16_t*)alloc((size_t)D_ * D_ * sizeof(bf16_t));
  bf16_t* wctx_b = (bf16_t*)alloc((size_t)D_ * D_ * sizeof(bf16_t));
  bf16_t* wx_b   = (bf16_t*)alloc((size_t)D_ * D_ * sizeof(bf16_t));
  bf16_t* ffw1_b = (bf16_t*)alloc((size_t)D_ * FF_ * sizeof(bf16_t));
  bf16_t* ffw2_b = (bf16_t*)alloc((size_t)FF_ * D_ * sizeof(bf16_t));
  // bf16 activations (reused across phases)
  bf16_t* bufA   = (bf16_t*)alloc(bufA_elems * sizeof(bf16_t)); // qkv -> ffn hidden
  bf16_t* actb   = (bf16_t*)alloc(ND * sizeof(bf16_t));         // xn -> ctxb -> hb
  bf16_t* ktxs   = (bf16_t*)alloc(ND * sizeof(bf16_t));         // K^T -> xs_bf16
  bf16_t* attnhb = (bf16_t*)alloc(ND * sizeof(bf16_t));         // attention heads
  // fp32 state
  float* apcpre = (float*)alloc(ND * sizeof(float));            // attnproj -> cpre
  float* x12    = (float*)alloc(ND * sizeof(float));            // x1 -> x2
  float* xs     = (float*)alloc(ND * sizeof(float));
  float* vs     = (float*)alloc(ND * sizeof(float));
  float* tmp    = (float*)alloc(ND * sizeof(float));

  auto gemm = [&](const bf16_t* A, const bf16_t* Bm, const float* bias,
                  const float* resid, float* oF, bf16_t* oB,
                  int M, int Nn, int K, int mode) {
    dim3 g(Nn / TN, M / TM);
    gemm_bf16_wmma<<<g, dim3(256), 0, stream>>>(A, Bm, bias, resid, oF, oB, M, Nn, K, mode);
  };
  auto nblk = [](size_t n) { return (unsigned)((n + 255) / 256); };

  // 0) weights -> bf16
  cast_f32_bf16<<<nblk((size_t)D_*3*D_), 256, 0, stream>>>(qkv_w, qkvw_b, D_*3*D_);
  cast_f32_bf16<<<nblk((size_t)D_*D_),   256, 0, stream>>>(out_w,  outw_b, D_*D_);
  cast_f32_bf16<<<nblk((size_t)D_*D_),   256, 0, stream>>>(w_ctx,  wctx_b, D_*D_);
  cast_f32_bf16<<<nblk((size_t)D_*D_),   256, 0, stream>>>(w_x,    wx_b,   D_*D_);
  cast_f32_bf16<<<nblk((size_t)D_*FF_),  256, 0, stream>>>(ff_w1,  ffw1_b, D_*FF_);
  cast_f32_bf16<<<nblk((size_t)FF_*D_),  256, 0, stream>>>(ff_w2,  ffw2_b, FF_*D_);

  // 1) xn = LN(x) -> bf16
  layernorm_f32<<<N_, 256, 0, stream>>>(x, g_attn, b_attn, (float*)nullptr, actb);
  // 2) qkv = xn @ qkv_w + qkv_b  (bf16 out)
  gemm(actb, qkvw_b, qkv_b, nullptr, nullptr, bufA, N_, 3 * D_, D_, 1);
  // 3) K^T buffer for attention
  transpose_k_kern<<<nblk(ND), 256, 0, stream>>>(bufA, ktxs);
  // 4) causal multi-head attention
  attn_causal_wmma<<<dim3(S_ / 16, B_ * H_), 32, 0, stream>>>(bufA, ktxs, attnhb);
  // 5) attnproj = heads @ out_w + out_b  (fp32)
  gemm(attnhb, outw_b, out_b, nullptr, apcpre, nullptr, N_, D_, D_, 0);
  // 6) x1 = x + attnproj ; ctxb = bf16(attnproj)
  resid_cast_kern<<<nblk(ND), 256, 0, stream>>>(x, apcpre, x12, actb, (int)ND);
  // 7) xs = LN(x1) (fp32 + bf16) ; vs = 0
  layernorm_f32<<<N_, 256, 0, stream>>>(x12, g1, b1, xs, ktxs);
  hipMemsetAsync(vs, 0, ND * sizeof(float), stream);
  // 8) cpre = ctx @ w_ctx + b_inl  (loop-invariant, hoisted)
  gemm(actb, wctx_b, b_inl, nullptr, apcpre, nullptr, N_, D_, D_, 0);
  // 9) 8 INL iterations: tmp = xs @ w_x ; semi-implicit Euler update
  for (int it = 0; it < 8; ++it) {
    gemm(ktxs, wx_b, nullptr, nullptr, tmp, nullptr, N_, D_, D_, 0);
    inl_update_kern<<<nblk(ND), 256, 0, stream>>>(apcpre, tmp, xs, vs, ktxs, (int)ND);
  }
  // 10) x2 = x1 + xs
  add_inplace_kern<<<nblk(ND), 256, 0, stream>>>(x12, xs, (int)ND);
  // 11) h = LN(x2) -> bf16
  layernorm_f32<<<N_, 256, 0, stream>>>(x12, g2, b2, (float*)nullptr, actb);
  // 12) h1 = gelu(h @ ff_w1 + ff_b1) -> bf16
  gemm(actb, ffw1_b, ff_b1, nullptr, nullptr, bufA, N_, FF_, D_, 2);
  // 13) out = x2 + h1 @ ff_w2 + ff_b2  (fp32)
  gemm(bufA, ffw2_b, ff_b2, x12, out, nullptr, N_, D_, FF_, 0);
}
